// GIN_78408922956190
// MI455X (gfx1250) — matrix operand order repaired and checked by reading, not compiled, not run on hardware
//
#include <hip/hip_runtime.h>

typedef __attribute__((ext_vector_type(16))) _Float16 v16h;
typedef __attribute__((ext_vector_type(8)))  _Float16 v8h;
typedef __attribute__((ext_vector_type(8)))  float    v8f;
typedef _Float16 half_t;

#define NODES   65536
#define DIM     256
#define GFEAT   64
#define NLAYER  4
#define MIX     10
#define BN_EPS  1e-5f

// ---------------------------------------------------------------------------
// Elementwise helpers
// ---------------------------------------------------------------------------
__global__ void k_zero(float* p, int n) {
    int i = blockIdx.x * blockDim.x + threadIdx.x;
    if (i < n) p[i] = 0.0f;
}

__global__ void k_f32_to_f16(const float* __restrict__ in, half_t* __restrict__ out, int n) {
    int i = blockIdx.x * blockDim.x + threadIdx.x;
    if (i < n) out[i] = (half_t)in[i];
}

__global__ void k_copy_f32x4(const float4* __restrict__ in, float4* __restrict__ out, int n4) {
    int i = blockIdx.x * blockDim.x + threadIdx.x;
    if (i < n4) out[i] = in[i];
}

// W is K x Nn (row-major, fp32). Wt is Nn x K (row-major, f16). Coalesced writes.
__global__ void k_transpose_w(const float* __restrict__ W, half_t* __restrict__ Wt,
                              int K, int Nn) {
    int i = blockIdx.x * blockDim.x + threadIdx.x;  // i = n*K + k
    if (i < K * Nn) {
        int n = i / K;
        int k = i - n * K;
        Wt[i] = (half_t)W[(size_t)k * Nn + n];
    }
}

// ---------------------------------------------------------------------------
// Edge scatter-add: z[dst] += h[src]   (z pre-initialized to h: self term)
// one block per edge, 256 threads = 256 channels
// ---------------------------------------------------------------------------
__global__ __launch_bounds__(256) void k_scatter(const int* __restrict__ ei, int E,
                                                 const float* __restrict__ h,
                                                 float* __restrict__ z) {
    int e = blockIdx.x;
    int c = threadIdx.x;
    int src = ei[e];
    int dst = ei[E + e];
    float v = h[(size_t)src * DIM + c];
    atomicAdd(&z[(size_t)dst * DIM + c], v);
}

// ---------------------------------------------------------------------------
// BatchNorm statistics (sum, sumsq per channel) via block partials + atomics
// grid = NODES/256 blocks, 256 threads (thread = channel)
// ---------------------------------------------------------------------------
__global__ __launch_bounds__(256) void k_bn_stats(const float* __restrict__ A,
                                                  float* __restrict__ stats) {
    int c = threadIdx.x;
    size_t row0 = (size_t)blockIdx.x * 256;
    float s = 0.0f, s2 = 0.0f;
    for (int r = 0; r < 256; ++r) {
        float v = A[(row0 + r) * DIM + c];
        s += v;
        s2 += v * v;
    }
    atomicAdd(&stats[c], s);
    atomicAdd(&stats[DIM + c], s2);
}

__global__ void k_bn_apply(float* __restrict__ A, const float* __restrict__ stats,
                           const float* __restrict__ g, const float* __restrict__ b,
                           int n) {
    int i = blockIdx.x * blockDim.x + threadIdx.x;
    if (i < n) {
        int c = i & (DIM - 1);
        float mean = stats[c] * (1.0f / NODES);
        float var  = stats[DIM + c] * (1.0f / NODES) - mean * mean;
        float inv  = rsqrtf(var + BN_EPS);
        A[i] = g[c] * (A[i] - mean) * inv + b[c];
    }
}

// ---------------------------------------------------------------------------
// WMMA GEMM:  Out[NODES x 256] = act( A[NODES x K] @ W[K x 256] + bias )
//   A : f16 row-major, Wt : f16, 256 rows x K cols (pre-transposed weights)
//   OUT_MODE 0: fp32 store (no activation)     -> float* out
//   OUT_MODE 1: relu, f16 store                -> half_t* out
//
// Block = 256 threads = 8 waves. blockIdx.y selects a 64-column group whose
// weight strip (64 x K f16, <= 32 KB) is staged once into LDS and shared by
// all 8 waves (B fragments via ds_load). Each wave computes a 32x64 strip:
// 2 A-frags (global, prefetched) x 4 B-frags (LDS) -> 8 WMMAs per K-step of 32,
// so each LDS B fragment feeds two WMMAs (1.5 loads per WMMA).
// ---------------------------------------------------------------------------
template <int OUT_MODE>
__global__ __launch_bounds__(256) void gemm_wmma(const half_t* __restrict__ A,
                                                 const half_t* __restrict__ Wt,
                                                 const float* __restrict__ bias,
                                                 void* __restrict__ out, int K) {
    __shared__ half_t bsh[64 * DIM];    // 32 KB max (64 cols x K halves)

    const int tid  = threadIdx.x;
    const int wave = tid >> 5;
    const int lane = tid & 31;
    const int n0   = blockIdx.y * 64;   // column group
    const int hi   = (lane >= 16) ? 1 : 0;
    const int lr   = lane & 15;

    // ---- stage weight strip Wt[n0 .. n0+63][0..K) into LDS (contiguous) ----
    const int nvec = (64 * K) / 8;      // v8h chunks
    const v8h* wsrc = (const v8h*)(Wt + (size_t)n0 * K);
    for (int i = tid; i < nvec; i += 256) ((v8h*)bsh)[i] = wsrc[i];
    __syncthreads();

    // ---- per-wave 32-row strip (two 16-row A tiles) ----
    const int m_base = (blockIdx.x * 8 + wave) * 32;
    const half_t* arow0 = A + (size_t)(m_base +      lr) * K + hi * 16;
    const half_t* arow1 = A + (size_t)(m_base + 16 + lr) * K + hi * 16;

    // B fragment sources in LDS: local col c*16+lr, 16 contiguous K-halves
    const half_t* bp0 = bsh + (size_t)( 0 + lr) * K + hi * 16;
    const half_t* bp1 = bsh + (size_t)(16 + lr) * K + hi * 16;
    const half_t* bp2 = bsh + (size_t)(32 + lr) * K + hi * 16;
    const half_t* bp3 = bsh + (size_t)(48 + lr) * K + hi * 16;

    v8f acc00 = {}, acc01 = {}, acc02 = {}, acc03 = {};
    v8f acc10 = {}, acc11 = {}, acc12 = {}, acc13 = {};

    for (int k0 = 0; k0 < K; k0 += 32) {
        __builtin_prefetch(arow0 + k0 + 64, 0, 0);   // global_prefetch_b8
        __builtin_prefetch(arow1 + k0 + 64, 0, 0);
        v8h a0lo = *(const v8h*)(arow0 + k0);
        v8h a0hi = *(const v8h*)(arow0 + k0 + 8);
        v8h a1lo = *(const v8h*)(arow1 + k0);
        v8h a1hi = *(const v8h*)(arow1 + k0 + 8);
        v16h a0 = __builtin_shufflevector(a0lo, a0hi, 0, 1, 2, 3, 4, 5, 6, 7,
                                                      8, 9, 10, 11, 12, 13, 14, 15);
        v16h a1 = __builtin_shufflevector(a1lo, a1hi, 0, 1, 2, 3, 4, 5, 6, 7,
                                                      8, 9, 10, 11, 12, 13, 14, 15);
        v16h b0 = *(const v16h*)(bp0 + k0);
        v16h b1 = *(const v16h*)(bp1 + k0);
        v16h b2 = *(const v16h*)(bp2 + k0);
        v16h b3 = *(const v16h*)(bp3 + k0);
        acc00 = __builtin_amdgcn_wmma_f32_16x16x32_f16(false, a0, false, b0, (short)0, acc00, false, false);
        acc10 = __builtin_amdgcn_wmma_f32_16x16x32_f16(false, a1, false, b0, (short)0, acc10, false, false);
        acc01 = __builtin_amdgcn_wmma_f32_16x16x32_f16(false, a0, false, b1, (short)0, acc01, false, false);
        acc11 = __builtin_amdgcn_wmma_f32_16x16x32_f16(false, a1, false, b1, (short)0, acc11, false, false);
        acc02 = __builtin_amdgcn_wmma_f32_16x16x32_f16(false, a0, false, b2, (short)0, acc02, false, false);
        acc12 = __builtin_amdgcn_wmma_f32_16x16x32_f16(false, a1, false, b2, (short)0, acc12, false, false);
        acc03 = __builtin_amdgcn_wmma_f32_16x16x32_f16(false, a0, false, b3, (short)0, acc03, false, false);
        acc13 = __builtin_amdgcn_wmma_f32_16x16x32_f16(false, a1, false, b3, (short)0, acc13, false, false);
    }

    // Epilogue. C/D layout: vgpr j -> row m0+j (lanes 0-15) or m0+8+j (lanes 16-31),
    // column = n_tile + (lane & 15).
    v8f accs[2][4] = {{acc00, acc01, acc02, acc03}, {acc10, acc11, acc12, acc13}};
#pragma unroll
    for (int r = 0; r < 2; ++r) {
        const int mrow0 = m_base + r * 16 + hi * 8;
#pragma unroll
        for (int c = 0; c < 4; ++c) {
            int n = n0 + c * 16 + lr;
            float bv = bias[n];
#pragma unroll
            for (int j = 0; j < 8; ++j) {
                float v = accs[r][c][j] + bv;
                if (OUT_MODE == 1) {
                    v = v > 0.0f ? v : 0.0f;
                    ((half_t*)out)[(size_t)(mrow0 + j) * DIM + n] = (half_t)v;
                } else {
                    ((float*)out)[(size_t)(mrow0 + j) * DIM + n] = v;
                }
            }
        }
    }
}

// ---------------------------------------------------------------------------
// Head final projection: y3[node] = dot(y2[node, :], w3) + b3   (D -> 1)
// ---------------------------------------------------------------------------
__global__ __launch_bounds__(256) void k_head_proj(const half_t* __restrict__ y2,
                                                   const float* __restrict__ w3,
                                                   const float* __restrict__ b3, int mh,
                                                   float* __restrict__ y3) {
    int node = blockIdx.x * blockDim.x + threadIdx.x;
    if (node < NODES) {
        const half_t* row = y2 + (size_t)node * DIM;
        float acc = 0.0f;
#pragma unroll 8
        for (int k = 0; k < DIM; ++k) acc += (float)row[k] * w3[k];
        y3[node] = acc + b3[mh];
    }
}

// mean / std (ddof=1) over MIX heads
__global__ __launch_bounds__(256) void k_mixture(const float* __restrict__ y3,
                                                 float* __restrict__ out) {
    int node = blockIdx.x * blockDim.x + threadIdx.x;
    if (node < NODES) {
        float v[MIX];
        float s = 0.0f;
#pragma unroll
        for (int mm = 0; mm < MIX; ++mm) { v[mm] = y3[(size_t)mm * NODES + node]; s += v[mm]; }
        float mu = s * (1.0f / MIX);
        float d2 = 0.0f;
#pragma unroll
        for (int mm = 0; mm < MIX; ++mm) { float d = v[mm] - mu; d2 += d * d; }
        out[node]         = mu;
        out[NODES + node] = sqrtf(d2 * (1.0f / (MIX - 1)));
    }
}

// ---------------------------------------------------------------------------
// Orchestration
// ---------------------------------------------------------------------------
static inline int cdiv(int a, int b) { return (a + b - 1) / b; }

extern "C" void kernel_launch(void* const* d_in, const int* in_sizes, int n_in,
                              void* d_out, int out_size, void* d_ws, size_t ws_size,
                              hipStream_t stream) {
    const float* x        = (const float*)d_in[0];
    const int*   ei       = (const int*)  d_in[1];
    const int    E        = in_sizes[1] / 2;
    const float* W_embed  = (const float*)d_in[3];
    const float* b_embed  = (const float*)d_in[4];
    const float* bn_in_g  = (const float*)d_in[5];
    const float* bn_in_b  = (const float*)d_in[6];
    const float* gin_W1   = (const float*)d_in[7];
    const float* gin_b1   = (const float*)d_in[8];
    const float* gin_W2   = (const float*)d_in[9];
    const float* gin_b2   = (const float*)d_in[10];
    const float* bn_g     = (const float*)d_in[11];
    const float* bn_b     = (const float*)d_in[12];
    const float* Wout1    = (const float*)d_in[13];
    const float* bout1    = (const float*)d_in[14];
    const float* Wout2    = (const float*)d_in[15];
    const float* bout2    = (const float*)d_in[16];
    const float* Wout3    = (const float*)d_in[17];
    const float* bout3    = (const float*)d_in[18];
    float* out = (float*)d_out;

    // workspace carve-up
    char* ws = (char*)d_ws;
    float*  h    = (float*)(ws);                                  // 64 MB
    float*  z    = (float*)(ws + (size_t)64 * 1024 * 1024);       // 64 MB
    half_t* f16a = (half_t*)(ws + (size_t)128 * 1024 * 1024);     // 32 MB
    half_t* f16b = (half_t*)(ws + (size_t)160 * 1024 * 1024);     // 32 MB
    half_t* wt1  = (half_t*)(ws + (size_t)192 * 1024 * 1024);     // 128 KB
    half_t* wt2  = (half_t*)(ws + (size_t)192 * 1024 * 1024 + 131072);
    float*  stats= (float*) (ws + (size_t)192 * 1024 * 1024 + 262144);  // 2*256 f32
    float*  y3   = (float*) (ws + (size_t)192 * 1024 * 1024 + 262144 + 4096);
    half_t* y2h  = (half_t*)z;   // z is dead during head phase

    const int ND = NODES * DIM;
    const dim3 ggrid(NODES / 32 / 8, DIM / 64);   // (256, 4)

    // ---- input embedding: h = BN(x @ W_embed + b) ----
    k_f32_to_f16<<<cdiv(NODES * GFEAT, 256), 256, 0, stream>>>(x, f16a, NODES * GFEAT);
    k_transpose_w<<<cdiv(GFEAT * DIM, 256), 256, 0, stream>>>(W_embed, wt1, GFEAT, DIM);
    gemm_wmma<0><<<ggrid, 256, 0, stream>>>(f16a, wt1, b_embed, h, GFEAT);
    k_zero<<<cdiv(2 * DIM, 256), 256, 0, stream>>>(stats, 2 * DIM);
    k_bn_stats<<<NODES / 256, 256, 0, stream>>>(h, stats);
    k_bn_apply<<<cdiv(ND, 256), 256, 0, stream>>>(h, stats, bn_in_g, bn_in_b, ND);

    // ---- 4 x GINConv ----
    for (int i = 0; i < NLAYER; ++i) {
        // z = h + segment_sum(h[src] -> dst)
        k_copy_f32x4<<<cdiv(ND / 4, 256), 256, 0, stream>>>((const float4*)h, (float4*)z, ND / 4);
        k_scatter<<<E, 256, 0, stream>>>(ei, E, h, z);
        // t = relu(z @ W1 + b1)   [f16]
        k_f32_to_f16<<<cdiv(ND, 256), 256, 0, stream>>>(z, f16a, ND);
        k_transpose_w<<<cdiv(DIM * DIM, 256), 256, 0, stream>>>(gin_W1 + (size_t)i * DIM * DIM, wt1, DIM, DIM);
        gemm_wmma<1><<<ggrid, 256, 0, stream>>>(f16a, wt1, gin_b1 + i * DIM, f16b, DIM);
        // h_pre = t @ W2 + b2     [fp32]
        k_transpose_w<<<cdiv(DIM * DIM, 256), 256, 0, stream>>>(gin_W2 + (size_t)i * DIM * DIM, wt2, DIM, DIM);
        gemm_wmma<0><<<ggrid, 256, 0, stream>>>(f16b, wt2, gin_b2 + i * DIM, h, DIM);
        // h = BN(h_pre)
        k_zero<<<cdiv(2 * DIM, 256), 256, 0, stream>>>(stats, 2 * DIM);
        k_bn_stats<<<NODES / 256, 256, 0, stream>>>(h, stats);
        k_bn_apply<<<cdiv(ND, 256), 256, 0, stream>>>(h, stats, bn_g + i * DIM, bn_b + i * DIM, ND);
    }

    // ---- mixture heads ----
    k_f32_to_f16<<<cdiv(ND, 256), 256, 0, stream>>>(h, f16a, ND);   // h_f16 persistent
    for (int mh = 0; mh < MIX; ++mh) {
        k_transpose_w<<<cdiv(DIM * DIM, 256), 256, 0, stream>>>(Wout1 + (size_t)mh * DIM * DIM, wt1, DIM, DIM);
        gemm_wmma<1><<<ggrid, 256, 0, stream>>>(f16a, wt1, bout1 + mh * DIM, f16b, DIM);
        k_transpose_w<<<cdiv(DIM * DIM, 256), 256, 0, stream>>>(Wout2 + (size_t)mh * DIM * DIM, wt2, DIM, DIM);
        gemm_wmma<1><<<ggrid, 256, 0, stream>>>(f16b, wt2, bout2 + mh * DIM, y2h, DIM);
        k_head_proj<<<NODES / 256, 256, 0, stream>>>(y2h, Wout3 + (size_t)mh * DIM, bout3, mh,
                                                     y3 + (size_t)mh * NODES);
    }

    // ---- bootstrap mean / std over heads ----
    k_mixture<<<NODES / 256, 256, 0, stream>>>(y3, out);
}
